// Reformer_Output_8083128451370
// MI455X (gfx1250) — matrix operand order repaired and checked by reading, not compiled, not run on hardware
//
#include <hip/hip_runtime.h>
#include <cmath>

namespace {

constexpr int Hdim   = 256;
constexpr int HEADSn = 8;
constexpr int DH     = 32;
constexpr int Bn     = 32;
constexpr int Nn     = 512;
constexpr int NHASH  = 8;
constexpr int NCHUNK = 256;          // N_HASHES * N / BUCKET
constexpr int BHn    = Bn * HEADSn;  // 256
constexpr int MTOK   = Bn * Nn;      // 16384
constexpr float SELF_VAL = -5.0e4f;

typedef _Float16 f16;
typedef _Float16 v8h  __attribute__((ext_vector_type(8)));
typedef _Float16 v16h __attribute__((ext_vector_type(16)));
typedef float    v8f  __attribute__((ext_vector_type(8)));
typedef int      v4i  __attribute__((vector_size(16)));  // matches builtin proto

#if defined(__has_builtin)
#if __has_builtin(__builtin_amdgcn_global_load_async_to_lds_b128)
#define USE_ASYNC 1
#else
#define USE_ASYNC 0
#endif
#else
#define USE_ASYNC 0
#endif

// 16-byte global->LDS copy: async (ASYNCcnt) when available, sync fallback.
__device__ inline void copy16(const f16* g, f16* l) {
#if USE_ASYNC
  __builtin_amdgcn_global_load_async_to_lds_b128(
      (__attribute__((address_space(1))) v4i*)(void*)const_cast<f16*>(g),
      (__attribute__((address_space(3))) v4i*)(void*)l, 0, 0);
#else
  *(v8h*)l = *(const v8h*)g;
#endif
}

template <int Ncnt>
__device__ inline void wait_async() {
#if USE_ASYNC
#if __has_builtin(__builtin_amdgcn_s_wait_asynccnt)
  __builtin_amdgcn_s_wait_asynccnt(Ncnt);
#else
  asm volatile("s_wait_asynccnt %0" ::"i"(Ncnt) : "memory");
#endif
#endif
}

// A-matrix fragment (16x32 f16): lane<16 -> M=lane, K in {0..7,16..23};
// lane>=16 -> M=lane-16, K in {8..15,24..31}. row = 32 contiguous halfs.
__device__ inline v16h frag_a(const f16* row, bool hi) {
  const v8h* p = (const v8h*)row;
  v8h x0 = p[hi ? 1 : 0];
  v8h x1 = p[hi ? 3 : 2];
  v16h r;
#pragma unroll
  for (int i = 0; i < 8; i++) { r[i] = x0[i]; r[8 + i] = x1[i]; }
  return r;
}
// B-matrix fragment (32x16 f16) from Bt[n][k] rows: lane<16 -> N=lane, K in {0..7,16..23};
// lane>=16 -> N=lane-16, K in {8..15,24..31}. row = 32 contiguous halfs over k.
__device__ inline v16h frag_b(const f16* row, bool hi) {
  const v8h* p = (const v8h*)row;
  v8h x0 = p[hi ? 1 : 0];
  v8h x1 = p[hi ? 3 : 2];
  v16h r;
#pragma unroll
  for (int i = 0; i < 8; i++) { r[i] = x0[i]; r[8 + i] = x1[i]; }
  return r;
}

__device__ inline float block_sum(float v, float* sh) {
  int tid = threadIdx.x;
  sh[tid] = v; __syncthreads();
  for (int s = 128; s > 0; s >>= 1) {
    if (tid < s) sh[tid] += sh[tid + s];
    __syncthreads();
  }
  float r = sh[0]; __syncthreads();
  return r;
}
__device__ inline float block_max(float v, float* sh) {
  int tid = threadIdx.x;
  sh[tid] = v; __syncthreads();
  for (int s = 128; s > 0; s >>= 1) {
    if (tid < s) sh[tid] = fmaxf(sh[tid], sh[tid + s]);
    __syncthreads();
  }
  float r = sh[0]; __syncthreads();
  return r;
}

// ---- token-wise dot products: l1 = att.w1+b1 + mod.wm1+bm1 ; l2att = att.w2+b2
__global__ void k_token_dots(const float* __restrict__ att, const float* __restrict__ mod,
                             const float* __restrict__ w1, const float* __restrict__ b1,
                             const float* __restrict__ wm1, const float* __restrict__ bm1,
                             const float* __restrict__ w2, const float* __restrict__ b2,
                             float* __restrict__ l1, float* __restrict__ l2att) {
  __shared__ float sh[256];
  int t = blockIdx.x, tid = threadIdx.x;
  const float* ar = att + (size_t)t * 1024;
  float s1 = 0.f, s2 = 0.f;
  for (int i = tid; i < 1024; i += 256) { float a = ar[i]; s1 += a * w1[i]; s2 += a * w2[i]; }
  float sm = mod[(size_t)t * Hdim + tid] * wm1[tid];
  s1 = block_sum(s1, sh);
  s2 = block_sum(s2, sh);
  sm = block_sum(sm, sh);
  if (tid == 0) { l1[t] = s1 + b1[0] + sm + bm1[0]; l2att[t] = s2 + b2[0]; }
}

// ---- layernorm over H=256, one block per token; f16 output (feeds WMMA GEMMs)
__global__ void k_layernorm(const float* __restrict__ x, const float* __restrict__ g,
                            const float* __restrict__ bta, f16* __restrict__ out) {
  __shared__ float sh[256];
  int t = blockIdx.x, tid = threadIdx.x;
  float v = x[(size_t)t * Hdim + tid];
  float mu = block_sum(v, sh) * (1.0f / Hdim);
  float dv = v - mu;
  float var = block_sum(dv * dv, sh) * (1.0f / Hdim);
  out[(size_t)t * Hdim + tid] = (f16)(dv * rsqrtf(var + 1e-5f) * g[tid] + bta[tid]);
}

// ---- one-time weight convert+transpose: Wt[n][k] = (f16)W[k][n]
__global__ void k_wt16(const float* __restrict__ W, f16* __restrict__ Wt, int K, int N) {
  int idx = blockIdx.x * 256 + threadIdx.x;  // over N*K
  int n = idx / K, k = idx % K;
  Wt[(size_t)n * K + k] = (f16)W[(size_t)k * N + n];
}

// ---- WMMA GEMM: out[M,N] = epi(A[M,K] @ W[K,N]); f16 A and Wt (pre-transposed).
// Block = 64 rows x 128 cols, 8 waves; each wave owns a 16x64 strip (4 wmma/chunk).
// Double-buffered LDS staging via async global->LDS b128 copies (3 per wave/chunk).
template <bool GELU, bool SCATTER, bool HALFOUT>
__global__ void k_gemm(const f16* __restrict__ A, const f16* __restrict__ Wt,
                       const float* __restrict__ bias, const float* __restrict__ res1,
                       const float* __restrict__ res2, void* __restrict__ outp,
                       int K, int Ncols, float scale) {
  alignas(16) __shared__ f16 As[2][64][32];
  alignas(16) __shared__ f16 Ws[2][128][32];
  int tid = threadIdx.x;
  int wv = tid >> 5, l = tid & 31, ln = l & 15;
  bool hi = l >= 16;
  int ms = wv >> 1, ng = wv & 1;  // wave -> (M strip 0..3, N half 0..1)
  int rowBase = blockIdx.x * 64;
  int colBase = blockIdx.y * 128;
  int nChunks = K / 32;

  auto stage = [&](int k0, f16(*as)[32], f16(*ws)[32]) {
    {
      int m = tid >> 2, ko = (tid & 3) * 8;  // 64 rows x 4 chunks
      copy16(A + (size_t)(rowBase + m) * K + k0 + ko, &as[m][ko]);
    }
#pragma unroll
    for (int e = 0; e < 2; e++) {
      int idx = e * 256 + tid;               // 128 rows x 4 chunks
      int n = idx >> 2, ko = (idx & 3) * 8;
      copy16(Wt + (size_t)(colBase + n) * K + k0 + ko, &ws[n][ko]);
    }
  };

  v8f acc[4] = {{}, {}, {}, {}};
  stage(0, As[0], Ws[0]);
  for (int c = 0; c < nChunks; c++) {
    int cur = c & 1;
    if (c + 1 < nChunks) {
      stage((c + 1) * 32, As[cur ^ 1], Ws[cur ^ 1]);
      wait_async<3>();  // drain older batch; keep the 3 just-issued in flight
    } else {
      wait_async<0>();
    }
    __syncthreads();
    v16h af = frag_a(&As[cur][ms * 16 + ln][0], hi);
#pragma unroll
    for (int j = 0; j < 4; j++) {
      v16h bf = frag_b(&Ws[cur][ng * 64 + j * 16 + ln][0], hi);
      acc[j] = __builtin_amdgcn_wmma_f32_16x16x32_f16(false, af, false, bf, (short)0,
                                                      acc[j], false, false);
    }
    __syncthreads();
  }

#pragma unroll
  for (int j = 0; j < 4; j++) {
    int col = colBase + ng * 64 + j * 16 + ln;
#pragma unroll
    for (int r = 0; r < 8; r++) {
      int row = rowBase + ms * 16 + r + (hi ? 8 : 0);
      float v = acc[j][r];
      if (bias) v += bias[col];
      if (GELU) v = 0.5f * v * (1.0f + erff(v * 0.70710678118654752f));
      size_t oi = (size_t)row * Ncols + col;
      if (res1) v += res1[oi];
      if (res2) v += res2[oi];
      v *= scale;
      if (SCATTER) {  // [B,N,H] col=(head,d) -> f32 [bh, n, dh]
        int b = row >> 9, n = row & 511;
        int head = col >> 5, d = col & 31;
        ((float*)outp)[(((size_t)b * HEADSn + head) * Nn + n) * DH + d] = v;
      } else if (HALFOUT) {
        ((f16*)outp)[oi] = (f16)v;
      } else {
        ((float*)outp)[oi] = v;
      }
    }
  }
}

// ---- LSH bucket assignment: argmax over [rot, -rot], first-max tie break
__global__ void k_hash(const float* __restrict__ qk, const float* __restrict__ rot,
                       int* __restrict__ buckets) {
  int gid = blockIdx.x * blockDim.x + threadIdx.x;  // < BH*N*NHASH
  int h = gid & 7;
  int n = (gid >> 3) & 511;
  int bh = gid >> 12;
  const float* q = qk + ((size_t)bh * Nn + n) * DH;
  float qr[32];
#pragma unroll
  for (int d = 0; d < 32; d++) qr[d] = q[d];
  float r16[16];
  for (int i = 0; i < 16; i++) {
    float s = 0.f;
    for (int d = 0; d < 32; d++) s += qr[d] * rot[(d * 8 + h) * 16 + i];
    r16[i] = s;
  }
  float best = -INFINITY; int bi = 0;
  for (int j = 0; j < 32; j++) {
    float v = (j < 16) ? r16[j] : -r16[j - 16];
    if (v > best) { best = v; bi = j; }
  }
  buckets[((size_t)bh * NHASH + h) * Nn + n] = bi;
}

// ---- stable counting sort per (bh, hash): 512 elems, 32 buckets, lane=bucket
__global__ void k_sort(const int* __restrict__ buckets, int* __restrict__ st) {
  __shared__ int sb[512];
  __shared__ int cnt[32];
  int lane = threadIdx.x;  // 32 threads
  int seg = blockIdx.x;    // bh*8 + h
  const int* bb = buckets + (size_t)seg * Nn;
  int* stb = st + (size_t)seg * Nn;
  for (int i = lane; i < Nn; i += 32) sb[i] = bb[i];
  __syncthreads();
  int c = 0;
  for (int i = 0; i < Nn; i++) if (sb[i] == lane) c++;
  cnt[lane] = c;
  __syncthreads();
  int start = 0;
  for (int j = 0; j < lane; j++) start += cnt[j];
  int k = 0;
  for (int i = 0; i < Nn; i++)
    if (sb[i] == lane) { stb[start + k] = i; k++; }
}

// ---- chunked LSH attention with look-one-back; scatters outputs by source pos
__global__ void k_attn(const float* __restrict__ qk, const float* __restrict__ vv,
                       const int* __restrict__ st,
                       float* __restrict__ o_all, float* __restrict__ lse_all) {
  alignas(16) __shared__ f16 qs[16][32];
  alignas(16) __shared__ f16 ks[32][32];   // normalized keys (== B^T for dots)
  alignas(16) __shared__ f16 vts[32][32];  // v transposed: vts[d][key]
  alignas(16) __shared__ f16 ps[16][32];   // probs as f16
  __shared__ float kf[32][32];
  __shared__ float dsm[16][32];
  __shared__ int tq[16];
  __shared__ int tk[32];

  int tid = threadIdx.x;  // 64 threads = 2 waves
  int bh = blockIdx.x / NCHUNK;
  int c = blockIdx.x % NCHUNK;
  int h = c / 32;
  int cprev = (c + NCHUNK - 1) % NCHUNK;  // global roll over all 256 chunks
  const int* stb = st + (size_t)bh * (NHASH * Nn);

  if (tid < 16) tq[tid] = stb[c * 16 + tid];
  if (tid < 32) tk[tid] = (tid < 16) ? stb[c * 16 + tid] : stb[cprev * 16 + (tid - 16)];
  __syncthreads();

  const float* qkb = qk + (size_t)bh * Nn * DH;
  const float* vb = vv + (size_t)bh * Nn * DH;
#pragma unroll
  for (int e = 0; e < 8; e++) {
    int lin = e * 64 + tid, r = lin >> 5, d = lin & 31;
    qs[r][d] = (f16)qkb[(size_t)tq[r] * DH + d];
  }
#pragma unroll
  for (int e = 0; e < 16; e++) {
    int lin = e * 64 + tid, j = lin >> 5, d = lin & 31;
    kf[j][d] = qkb[(size_t)tk[j] * DH + d];
    vts[d][j] = (f16)vb[(size_t)tk[j] * DH + d];
  }
  __syncthreads();
  if (tid < 32) {  // normalize keys: bk = bqk / max(||bqk||, 1e-12)
    float s = 0.f;
    for (int d = 0; d < 32; d++) { float x = kf[tid][d]; s += x * x; }
    float inv = 1.0f / fmaxf(sqrtf(s), 1e-12f);
    for (int d = 0; d < 32; d++) ks[tid][d] = (f16)(kf[tid][d] * inv);
  }
  __syncthreads();

  int wv = tid >> 5, l = tid & 31, ln = l & 15;
  bool hi = l >= 16;
  {  // dots = bq @ bk^T * DH^-0.5, self-mask
    v16h af = frag_a(&qs[ln][0], hi);
    v16h bf = frag_b(&ks[wv * 16 + ln][0], hi);
    v8f acc = {};
    acc = __builtin_amdgcn_wmma_f32_16x16x32_f16(false, af, false, bf, (short)0, acc, false, false);
#pragma unroll
    for (int r = 0; r < 8; r++) {
      int m = r + (hi ? 8 : 0);
      int col = wv * 16 + ln;
      float v = acc[r] * 0.17677669529663688f;  // 1/sqrt(32)
      if (tq[m] == tk[col]) v = SELF_VAL;
      dsm[m][col] = v;
    }
  }
  __syncthreads();
  if (tid < 16) {  // per-row LSE + probs
    float mx = -INFINITY;
    for (int j = 0; j < 32; j++) mx = fmaxf(mx, dsm[tid][j]);
    float s = 0.f;
    for (int j = 0; j < 32; j++) s += expf(dsm[tid][j] - mx);
    float lse = mx + logf(s);
    for (int j = 0; j < 32; j++) ps[tid][j] = (f16)expf(dsm[tid][j] - lse);
    lse_all[((size_t)bh * NHASH + h) * Nn + tq[tid]] = lse;
  }
  __syncthreads();
  {  // bo = probs @ bv, scatter by source position (implicit unsort)
    v16h af = frag_a(&ps[ln][0], hi);
    v16h bf = frag_b(&vts[wv * 16 + ln][0], hi);
    v8f acc = {};
    acc = __builtin_amdgcn_wmma_f32_16x16x32_f16(false, af, false, bf, (short)0, acc, false, false);
#pragma unroll
    for (int r = 0; r < 8; r++) {
      int m = r + (hi ? 8 : 0);
      int d = wv * 16 + ln;
      o_all[(((size_t)bh * NHASH + h) * Nn + tq[m]) * DH + d] = acc[r];
    }
  }
}

// ---- combine 8 hash rounds: out = sum_h o_h * softmax_h(lse); f16 out (GEMM input)
__global__ void k_combine(const float* __restrict__ o_all, const float* __restrict__ lse_all,
                          f16* __restrict__ attn) {
  int gid = blockIdx.x * blockDim.x + threadIdx.x;  // < BH*N*DH
  int d = gid & 31;
  int p = (gid >> 5) & 511;
  int bh = gid >> 14;
  int b = bh >> 3, head = bh & 7;
  float lv[8];
  float mx = -INFINITY;
  for (int h = 0; h < 8; h++) {
    lv[h] = lse_all[((size_t)bh * NHASH + h) * Nn + p];
    mx = fmaxf(mx, lv[h]);
  }
  float den = 0.f, num = 0.f;
  for (int h = 0; h < 8; h++) {
    float e = expf(lv[h] - mx);
    den += e;
    num += e * o_all[(((size_t)bh * NHASH + h) * Nn + p) * DH + d];
  }
  attn[((size_t)b * Nn + p) * Hdim + head * DH + d] = (f16)(num / den);
}

// ---- logits_2 += mod2 . w_mod2 + b_mod2
__global__ void k_logits2(const float* __restrict__ mod2, const float* __restrict__ wm2,
                          const float* __restrict__ bm2, float* __restrict__ l2) {
  __shared__ float sh[256];
  int t = blockIdx.x, tid = threadIdx.x;
  float s = mod2[(size_t)t * Hdim + tid] * wm2[tid];
  s = block_sum(s, sh);
  if (tid == 0) l2[t] = l2[t] + s + bm2[0];
}

// ---- masked log-softmax over N=512, one block per (batch, which-output)
__global__ void k_logsoftmax(const float* __restrict__ l1, const float* __restrict__ l2,
                             const int* __restrict__ mask, float* __restrict__ out) {
  __shared__ float sh[256];
  int b = blockIdx.x, o = blockIdx.y, tid = threadIdx.x;
  const float* src = o ? l2 : l1;
  int j0 = tid, j1 = tid + 256;
  float z0 = mask[b * Nn + j0] ? src[b * Nn + j0] : -1e30f;
  float z1 = mask[b * Nn + j1] ? src[b * Nn + j1] : -1e30f;
  float mx = block_max(fmaxf(z0, z1), sh);
  float s = block_sum(expf(z0 - mx) + expf(z1 - mx), sh);
  float lse = mx + logf(s);
  out[(size_t)o * MTOK + b * Nn + j0] = z0 - lse;
  out[(size_t)o * MTOK + b * Nn + j1] = z1 - lse;
}

}  // namespace

extern "C" void kernel_launch(void* const* d_in, const int* in_sizes, int n_in,
                              void* d_out, int out_size, void* d_ws, size_t ws_size,
                              hipStream_t stream) {
  const float* att    = (const float*)d_in[0];
  const float* mod    = (const float*)d_in[1];
  const int*   mask   = (const int*)d_in[2];
  const float* w_att1 = (const float*)d_in[3];
  const float* b_att1 = (const float*)d_in[4];
  const float* w_mod1 = (const float*)d_in[5];
  const float* b_mod1 = (const float*)d_in[6];
  const float* w_att2 = (const float*)d_in[7];
  const float* b_att2 = (const float*)d_in[8];
  const float* w_mod2 = (const float*)d_in[9];
  const float* b_mod2 = (const float*)d_in[10];
  const float* ln1_g  = (const float*)d_in[11];
  const float* ln1_b  = (const float*)d_in[12];
  const float* w_qk   = (const float*)d_in[13];
  const float* w_v    = (const float*)d_in[14];
  const float* w_out  = (const float*)d_in[15];
  const float* b_out  = (const float*)d_in[16];
  const float* ln2_g  = (const float*)d_in[17];
  const float* ln2_b  = (const float*)d_in[18];
  const float* w_ff1  = (const float*)d_in[19];
  const float* b_ff1  = (const float*)d_in[20];
  const float* w_ff2  = (const float*)d_in[21];
  const float* b_ff2  = (const float*)d_in[22];
  const float* rotations = (const float*)d_in[23];
  float* out = (float*)d_out;

  char* base = (char*)d_ws;
  size_t off = 0;
  auto alloc = [&](size_t bytes) -> void* {
    void* p = base + off;
    off += (bytes + 255) & ~(size_t)255;
    return p;
  };
  f16*   xln  = (f16*)  alloc((size_t)MTOK * Hdim * 2);
  float* qkh  = (float*)alloc((size_t)BHn * Nn * DH * 4);
  float* vh   = (float*)alloc((size_t)BHn * Nn * DH * 4);
  int*   bkts = (int*)  alloc((size_t)BHn * NHASH * Nn * 4);
  int*   st   = (int*)  alloc((size_t)BHn * NHASH * Nn * 4);
  float* lseA = (float*)alloc((size_t)BHn * NHASH * Nn * 4);
  float* oA   = (float*)alloc((size_t)BHn * NHASH * Nn * DH * 4);
  f16*   atc  = (f16*)  alloc((size_t)MTOK * Hdim * 2);
  float* y1   = (float*)alloc((size_t)MTOK * Hdim * 4);
  f16*   hln2 = (f16*)  alloc((size_t)MTOK * Hdim * 2);
  f16*   ffh  = (f16*)  alloc((size_t)MTOK * 4 * Hdim * 2);
  float* mod2 = (float*)alloc((size_t)MTOK * Hdim * 4);
  float* l1r  = (float*)alloc((size_t)MTOK * 4);
  float* l2r  = (float*)alloc((size_t)MTOK * 4);
  f16* wqk_t  = (f16*)alloc((size_t)Hdim * Hdim * 2);
  f16* wv_t   = (f16*)alloc((size_t)Hdim * Hdim * 2);
  f16* wout_t = (f16*)alloc((size_t)Hdim * Hdim * 2);
  f16* wff1_t = (f16*)alloc((size_t)Hdim * 4 * Hdim * 2);
  f16* wff2_t = (f16*)alloc((size_t)Hdim * 4 * Hdim * 2);

  // one-time weight f16 transposes (L2-resident afterwards)
  k_wt16<<<(Hdim * Hdim) / 256, 256, 0, stream>>>(w_qk, wqk_t, Hdim, Hdim);
  k_wt16<<<(Hdim * Hdim) / 256, 256, 0, stream>>>(w_v, wv_t, Hdim, Hdim);
  k_wt16<<<(Hdim * Hdim) / 256, 256, 0, stream>>>(w_out, wout_t, Hdim, Hdim);
  k_wt16<<<(Hdim * 4 * Hdim) / 256, 256, 0, stream>>>(w_ff1, wff1_t, Hdim, 4 * Hdim);
  k_wt16<<<(Hdim * 4 * Hdim) / 256, 256, 0, stream>>>(w_ff2, wff2_t, 4 * Hdim, Hdim);

  k_token_dots<<<MTOK, 256, 0, stream>>>(att, mod, w_att1, b_att1, w_mod1, b_mod1,
                                         w_att2, b_att2, l1r, l2r);
  k_layernorm<<<MTOK, 256, 0, stream>>>(mod, ln1_g, ln1_b, xln);
  k_gemm<false, true, false><<<dim3(MTOK / 64, Hdim / 128), 256, 0, stream>>>(
      xln, wqk_t, nullptr, nullptr, nullptr, qkh, Hdim, Hdim, 1.0f);
  k_gemm<false, true, false><<<dim3(MTOK / 64, Hdim / 128), 256, 0, stream>>>(
      xln, wv_t, nullptr, nullptr, nullptr, vh, Hdim, Hdim, 1.0f);
  k_hash<<<(BHn * Nn * NHASH) / 256, 256, 0, stream>>>(qkh, rotations, bkts);
  k_sort<<<BHn * NHASH, 32, 0, stream>>>(bkts, st);
  k_attn<<<BHn * NCHUNK, 64, 0, stream>>>(qkh, vh, st, oA, lseA);
  k_combine<<<(BHn * Nn * DH) / 256, 256, 0, stream>>>(oA, lseA, atc);
  k_gemm<false, false, false><<<dim3(MTOK / 64, Hdim / 128), 256, 0, stream>>>(
      atc, wout_t, b_out, mod, nullptr, y1, Hdim, Hdim, 1.0f);
  k_layernorm<<<MTOK, 256, 0, stream>>>(y1, ln2_g, ln2_b, hln2);
  k_gemm<true, false, true><<<dim3(MTOK / 64, (4 * Hdim) / 128), 256, 0, stream>>>(
      hln2, wff1_t, b_ff1, nullptr, nullptr, ffh, Hdim, 4 * Hdim, 1.0f);
  k_gemm<false, false, false><<<dim3(MTOK / 64, Hdim / 128), 256, 0, stream>>>(
      ffh, wff2_t, b_ff2, y1, mod, mod2, 4 * Hdim, Hdim, 0.5f);
  k_logits2<<<MTOK, 256, 0, stream>>>(mod2, w_mod2, b_mod2, l2r);
  k_logsoftmax<<<dim3(Bn, 2), 256, 0, stream>>>(l1r, l2r, mask, out);

  (void)in_sizes; (void)n_in; (void)out_size; (void)ws_size;
}
